// MSA_8306466751234
// MI455X (gfx1250) — matrix operand order repaired and checked by reading, not compiled, not run on hardware
//
#include <hip/hip_runtime.h>
#include <hip/hip_bf16.h>

// ---------------------------------------------------------------------------
// Multi-head self-attention for MI455X (gfx1250), wave32 + WMMA f16.
//   B=4 T=12 N=512 D=512 NH=8 DH=64 ; SCALE = 1/sqrt(DH/NH) = 1/sqrt(8)
// Pipeline: qkv_gemm (f32->f16, WMMA) -> attn (WMMA+softmax) -> out_gemm.
// Workspace layout (f16):
//   q_ws  [384][512][64]
//   k_ws  [384][512][64]
//   vT_ws [384][64][512]   (V stored transposed so PV B-frags are contiguous)
//   O_all [48][512][512]
// Total ws = 4 * 12,582,912 * 2 B = 100,663,296 bytes.
// ---------------------------------------------------------------------------

typedef _Float16 h16_t;
typedef __attribute__((ext_vector_type(16))) _Float16 v16h;
typedef __attribute__((ext_vector_type(8)))  float    v8f;
typedef __attribute__((ext_vector_type(4)))  unsigned int u32x4;
typedef __attribute__((ext_vector_type(4)))  float    f32x4;

union FragH  { v16h v; u32x4 q[2]; h16_t h[16]; };
union FragC  { v8f  v; float f[8]; };
union PackH2 { h16_t h[2]; unsigned int u; };

__device__ __forceinline__ v8f wmma_f16(v16h a, v16h b, v8f c) {
  // D = A(16x32 f16) * B(32x16 f16) + C(16x16 f32)
  return __builtin_amdgcn_wmma_f32_16x16x32_f16(
      /*neg_a=*/false, a, /*neg_b=*/false, b,
      /*c_mod=*/(short)0, c, /*reuse_a=*/false, /*reuse_b=*/false);
}

__device__ __forceinline__ v8f zero8() { v8f z = {0.f,0.f,0.f,0.f,0.f,0.f,0.f,0.f}; return z; }

#define GM_M   24576   // B*T*N rows
#define DMODEL 512
#define NQKV   1536
#define NHEADS 8
#define DH     64
#define NSEQ   512
#define NBTH   384     // B*T*NH
#define SCALE_ATTN 0.3535533905932738f   // 1/sqrt(8)  (reference quirk)
#define NEG_MASK  -1000000000.0f
#define LDSTR  40      // LDS row stride (elements), 80B -> 16B-aligned chunks

// ---------------------------------------------------------------------------
// Kernel 1: qkv = X @ wx_w^T + wx_b ; scatter into q/k/vT (f16) head-major.
// Block 256 thr (8 waves); tile 128(M)x128(N); K chunk 32; wave tile 32x64.
// ---------------------------------------------------------------------------
__global__ __launch_bounds__(256) void qkv_gemm_kernel(
    const float* __restrict__ X,    // [24576,512]
    const float* __restrict__ Wx,   // [1536,512]
    const float* __restrict__ bx,   // [1536]
    h16_t* __restrict__ q_ws, h16_t* __restrict__ k_ws, h16_t* __restrict__ vT_ws)
{
  __shared__ h16_t sA[128][LDSTR];
  __shared__ h16_t sB[128][LDSTR];
  const int tid  = threadIdx.x;
  const int nb0  = blockIdx.x * 128;
  const int mb0  = blockIdx.y * 128;
  const int w    = tid >> 5, lane = tid & 31;
  const int wm   = w >> 1,   wn   = w & 1;
  const int lr   = lane & 15;
  const int hs   = (lane < 16) ? 0 : 1;

  FragC acc[2][4];
#pragma unroll
  for (int mi = 0; mi < 2; ++mi)
#pragma unroll
    for (int ni = 0; ni < 4; ++ni) acc[mi][ni].v = zero8();

  for (int k0 = 0; k0 < DMODEL; k0 += 32) {
    // Stage A (128x32 f32 -> f16) and B (128x32) into LDS, packed uint stores.
#pragma unroll
    for (int it = 0; it < 4; ++it) {
      int f = tid + it * 256;            // float4 id 0..1023
      int row = f >> 3, col = (f & 7) * 4;
      f32x4 s = *(const f32x4*)(X + (size_t)(mb0 + row) * DMODEL + k0 + col);
      PackH2 p0; p0.h[0] = (h16_t)s.x; p0.h[1] = (h16_t)s.y;
      PackH2 p1; p1.h[0] = (h16_t)s.z; p1.h[1] = (h16_t)s.w;
      *(unsigned int*)&sA[row][col]     = p0.u;
      *(unsigned int*)&sA[row][col + 2] = p1.u;
    }
#pragma unroll
    for (int it = 0; it < 4; ++it) {
      int f = tid + it * 256;
      int row = f >> 3, col = (f & 7) * 4;
      f32x4 s = *(const f32x4*)(Wx + (size_t)(nb0 + row) * DMODEL + k0 + col);
      PackH2 p0; p0.h[0] = (h16_t)s.x; p0.h[1] = (h16_t)s.y;
      PackH2 p1; p1.h[0] = (h16_t)s.z; p1.h[1] = (h16_t)s.w;
      *(unsigned int*)&sB[row][col]     = p0.u;
      *(unsigned int*)&sB[row][col + 2] = p1.u;
    }
    __syncthreads();

    FragH a[2], b[4];
#pragma unroll
    for (int mi = 0; mi < 2; ++mi) {       // A 16x32: lane L row L; K {0..7,16..23}/{8..15,24..31}
      int r = wm * 32 + mi * 16 + lr;
      int kc = hs * 8;
      a[mi].q[0] = *(const u32x4*)&sA[r][kc];
      a[mi].q[1] = *(const u32x4*)&sA[r][kc + 16];
    }
#pragma unroll
    for (int ni = 0; ni < 4; ++ni) {       // B 32x16: lane c col c; K {0..7,16..23}/{8..15,24..31}
      int r = wn * 64 + ni * 16 + lr;
      int kc = hs * 8;
      b[ni].q[0] = *(const u32x4*)&sB[r][kc];
      b[ni].q[1] = *(const u32x4*)&sB[r][kc + 16];
    }
#pragma unroll
    for (int mi = 0; mi < 2; ++mi)
#pragma unroll
      for (int ni = 0; ni < 4; ++ni)
        acc[mi][ni].v = wmma_f16(a[mi].v, b[ni].v, acc[mi][ni].v);
    __syncthreads();
  }

  // Scatter C (+bias) to q / k / vT in f16.
#pragma unroll
  for (int mi = 0; mi < 2; ++mi) {
#pragma unroll
    for (int ni = 0; ni < 4; ++ni) {
      int gcol = nb0 + wn * 64 + ni * 16 + lr;
      float bias = bx[gcol];
      int part = gcol >> 9;          // 0=q 1=k 2=v
      int c = gcol & 511;
      int h = c >> 6, d = c & 63;
#pragma unroll
      for (int e = 0; e < 8; ++e) {
        int gm = mb0 + wm * 32 + mi * 16 + e + hs * 8;
        int bt = gm >> 9, n = gm & 511;
        int bth = bt * NHEADS + h;
        h16_t val = (h16_t)(acc[mi][ni].f[e] + bias);
        if (part == 0)      q_ws[((size_t)bth * NSEQ + n) * DH + d] = val;
        else if (part == 1) k_ws[((size_t)bth * NSEQ + n) * DH + d] = val;
        else                vT_ws[((size_t)bth * DH + d) * NSEQ + n] = val;
      }
    }
  }
}

// ---------------------------------------------------------------------------
// Kernel 2: per (bth, 16-row q tile): S = scale*q k^T + M*NEG; softmax; O=P V.
// Block 128 thr (4 waves). Wave w: score cols [w*128, w*128+128), then d-tile
// [w*16, w*16+16) of O. Scores live in LDS f32; P in LDS f16.
// ---------------------------------------------------------------------------
__global__ __launch_bounds__(128) void attn_kernel(
    const h16_t* __restrict__ q_ws, const h16_t* __restrict__ k_ws,
    const h16_t* __restrict__ vT_ws, const int* __restrict__ Mmask,
    h16_t* __restrict__ O_all)
{
  __shared__ float sS[16][520];
  __shared__ h16_t sP[16][520];
  const int tid  = threadIdx.x;
  const int w    = tid >> 5, lane = tid & 31;
  const int lr   = lane & 15;
  const int hs   = (lane < 16) ? 0 : 1;
  const int qt   = blockIdx.x;           // 0..31
  const int bth  = blockIdx.y;           // 0..383
  const int bt   = bth >> 3, h = bth & 7;
  const int ibase = qt * 16;

  // Q fragments for this 16-row tile, K(=DH) split 0..31 / 32..63.
  FragH aq[2];
  {
    const h16_t* qrow = q_ws + ((size_t)bth * NSEQ + ibase + lr) * DH;
#pragma unroll
    for (int kc = 0; kc < 2; ++kc) {
      int d0 = kc * 32 + hs * 8;
      aq[kc].q[0] = *(const u32x4*)(qrow + d0);
      aq[kc].q[1] = *(const u32x4*)(qrow + d0 + 16);
    }
  }

  // Phase A: scores into LDS.
  for (int jt = 0; jt < 8; ++jt) {
    const int j0 = w * 128 + jt * 16;
    FragH bk[2];
    const h16_t* krow = k_ws + ((size_t)bth * NSEQ + j0 + lr) * DH;
#pragma unroll
    for (int kc = 0; kc < 2; ++kc) {
      int d0 = kc * 32 + hs * 8;
      bk[kc].q[0] = *(const u32x4*)(krow + d0);
      bk[kc].q[1] = *(const u32x4*)(krow + d0 + 16);
    }
    FragC acc; acc.v = zero8();
    acc.v = wmma_f16(aq[0].v, bk[0].v, acc.v);
    acc.v = wmma_f16(aq[1].v, bk[1].v, acc.v);
    const int j = j0 + lr;
#pragma unroll
    for (int e = 0; e < 8; ++e) {
      int il = e + hs * 8;
      int gi = ibase + il;
      float m = (float)Mmask[((size_t)bt * NSEQ + gi) * NSEQ + j];
      sS[il][j] = acc.f[e] * SCALE_ATTN + m * NEG_MASK;
    }
  }
  __syncthreads();

  // Phase B: softmax over each 512-wide row; 8 consecutive lanes per row.
  {
    const int row = tid >> 3, sub = tid & 7;
    float mx = -3.4e38f;
    for (int j = sub; j < NSEQ; j += 8) mx = fmaxf(mx, sS[row][j]);
#pragma unroll
    for (int o = 4; o >= 1; o >>= 1) mx = fmaxf(mx, __shfl_xor(mx, o, 8));
    float sum = 0.f;
    for (int j = sub; j < NSEQ; j += 8) {
      float e = __expf(sS[row][j] - mx);
      sS[row][j] = e; sum += e;
    }
#pragma unroll
    for (int o = 4; o >= 1; o >>= 1) sum += __shfl_xor(sum, o, 8);
    float inv = 1.0f / sum;
    for (int j = sub; j < NSEQ; j += 8) sP[row][j] = (h16_t)(sS[row][j] * inv);
  }
  __syncthreads();

  // Phase C: O(16 x 16-dtile) = P(16x512) @ V(512x64); V read transposed.
  FragC o; o.v = zero8();
  const int dbase = w * 16;
  for (int j0 = 0; j0 < NSEQ; j0 += 32) {
    FragH a, b;
    int kA = j0 + hs * 8;
    a.q[0] = *(const u32x4*)&sP[lr][kA];
    a.q[1] = *(const u32x4*)&sP[lr][kA + 16];
    const h16_t* vrow = vT_ws + ((size_t)bth * DH + dbase + lr) * NSEQ + j0 + hs * 8;
    b.q[0] = *(const u32x4*)(vrow);
    b.q[1] = *(const u32x4*)(vrow + 16);
    o.v = wmma_f16(a.v, b.v, o.v);
  }
#pragma unroll
  for (int e = 0; e < 8; ++e) {
    int gi = ibase + e + hs * 8;
    O_all[((size_t)bt * NSEQ + gi) * DMODEL + h * DH + dbase + lr] = (h16_t)o.f[e];
  }
}

// ---------------------------------------------------------------------------
// Kernel 3: out = O_all(f16) @ wo_w^T + wo_b  ->  f32.  Same tiling as k1.
// ---------------------------------------------------------------------------
__global__ __launch_bounds__(256) void out_gemm_kernel(
    const h16_t* __restrict__ Oall,  // [24576,512] f16
    const float* __restrict__ Wo,    // [512,512]
    const float* __restrict__ bo,    // [512]
    float* __restrict__ out)         // [24576,512]
{
  __shared__ h16_t sA[128][LDSTR];
  __shared__ h16_t sB[128][LDSTR];
  const int tid = threadIdx.x;
  const int nb0 = blockIdx.x * 128;
  const int mb0 = blockIdx.y * 128;
  const int w   = tid >> 5, lane = tid & 31;
  const int wm  = w >> 1,   wn   = w & 1;
  const int lr  = lane & 15;
  const int hs  = (lane < 16) ? 0 : 1;

  FragC acc[2][4];
#pragma unroll
  for (int mi = 0; mi < 2; ++mi)
#pragma unroll
    for (int ni = 0; ni < 4; ++ni) acc[mi][ni].v = zero8();

  for (int k0 = 0; k0 < DMODEL; k0 += 32) {
    // A is already f16: straight 128-bit copies global -> LDS.
#pragma unroll
    for (int it = 0; it < 2; ++it) {
      int f = tid + it * 256;            // u32x4 id 0..511
      int row = f >> 2, c8 = (f & 3) * 8;
      *(u32x4*)&sA[row][c8] =
          *(const u32x4*)(Oall + (size_t)(mb0 + row) * DMODEL + k0 + c8);
    }
#pragma unroll
    for (int it = 0; it < 4; ++it) {
      int f = tid + it * 256;
      int row = f >> 3, col = (f & 7) * 4;
      f32x4 s = *(const f32x4*)(Wo + (size_t)(nb0 + row) * DMODEL + k0 + col);
      PackH2 p0; p0.h[0] = (h16_t)s.x; p0.h[1] = (h16_t)s.y;
      PackH2 p1; p1.h[0] = (h16_t)s.z; p1.h[1] = (h16_t)s.w;
      *(unsigned int*)&sB[row][col]     = p0.u;
      *(unsigned int*)&sB[row][col + 2] = p1.u;
    }
    __syncthreads();

    FragH a[2], b[4];
#pragma unroll
    for (int mi = 0; mi < 2; ++mi) {
      int r = wm * 32 + mi * 16 + lr;
      int kc = hs * 8;
      a[mi].q[0] = *(const u32x4*)&sA[r][kc];
      a[mi].q[1] = *(const u32x4*)&sA[r][kc + 16];
    }
#pragma unroll
    for (int ni = 0; ni < 4; ++ni) {
      int r = wn * 64 + ni * 16 + lr;
      int kc = hs * 8;
      b[ni].q[0] = *(const u32x4*)&sB[r][kc];
      b[ni].q[1] = *(const u32x4*)&sB[r][kc + 16];
    }
#pragma unroll
    for (int mi = 0; mi < 2; ++mi)
#pragma unroll
      for (int ni = 0; ni < 4; ++ni)
        acc[mi][ni].v = wmma_f16(a[mi].v, b[ni].v, acc[mi][ni].v);
    __syncthreads();
  }

#pragma unroll
  for (int mi = 0; mi < 2; ++mi) {
#pragma unroll
    for (int ni = 0; ni < 4; ++ni) {
      int gcol = nb0 + wn * 64 + ni * 16 + lr;
      float bias = bo[gcol];
#pragma unroll
      for (int e = 0; e < 8; ++e) {
        int gm = mb0 + wm * 32 + mi * 16 + e + hs * 8;
        out[(size_t)gm * DMODEL + gcol] = acc[mi][ni].f[e] + bias;
      }
    }
  }
}

// ---------------------------------------------------------------------------
extern "C" void kernel_launch(void* const* d_in, const int* in_sizes, int n_in,
                              void* d_out, int out_size, void* d_ws, size_t ws_size,
                              hipStream_t stream) {
  // setup_inputs order: V(0) K(1) Q(2) M(3) wx_w(4) wx_b(5) wo_w(6) wo_b(7).
  // Reference's self-attention branch ignores inputs V and K.
  (void)in_sizes; (void)n_in; (void)out_size; (void)ws_size;
  const float* X    = (const float*)d_in[2];
  const int*   Mm   = (const int*)d_in[3];
  const float* wx_w = (const float*)d_in[4];
  const float* wx_b = (const float*)d_in[5];
  const float* wo_w = (const float*)d_in[6];
  const float* wo_b = (const float*)d_in[7];
  float* out = (float*)d_out;

  const size_t HEAD_ELEMS = (size_t)NBTH * NSEQ * DH;   // 12,582,912
  h16_t* q_ws  = (h16_t*)d_ws;
  h16_t* k_ws  = q_ws  + HEAD_ELEMS;
  h16_t* vT_ws = k_ws  + HEAD_ELEMS;
  h16_t* O_all = vT_ws + HEAD_ELEMS;                    // ws total ~96 MiB

  qkv_gemm_kernel<<<dim3(NQKV / 128, GM_M / 128), 256, 0, stream>>>(
      X, wx_w, wx_b, q_ws, k_ws, vT_ws);
  attn_kernel<<<dim3(NSEQ / 16, NBTH), 128, 0, stream>>>(
      q_ws, k_ws, vT_ws, Mm, O_all);
  out_gemm_kernel<<<dim3(DMODEL / 128, GM_M / 128), 256, 0, stream>>>(
      O_all, wo_w, wo_b, out);
}
